// MTFE_19894288515306
// MI455X (gfx1250) — matrix operand order repaired and checked by reading, not compiled, not run on hardware
//
#include <hip/hip_runtime.h>
#include <math.h>

// ---------------------------------------------------------------------------
// Problem constants (fixed by the reference)
// ---------------------------------------------------------------------------
static constexpr int  kB   = 8;
static constexpr int  kC   = 64;     // channels (== CIN)
static constexpr int  kH   = 62;
static constexpr int  kT   = 400;
static constexpr int  kS   = 32;     // SSM state
static constexpr int  kMID = 4;      // channel-attn bottleneck
static constexpr int  kHT  = kH * kT;            // 24800  (divisible by 16)
static constexpr int  kNPOS = kB * kHT;          // 198400 (= 775*256)
static constexpr int  kPT  = kHT / 16;           // 1550 position tiles per batch
static constexpr long long kNELEM = (long long)kC * kNPOS; // 12,697,600 (= 49600*256)
static constexpr int  kWSLOT = 4096;             // f16 elems per packed weight

typedef _Float16 f16;
typedef __attribute__((ext_vector_type(16))) _Float16 v16h;
typedef __attribute__((ext_vector_type(8)))  float    v8f;

// Native-rate transcendentals (v_exp_f32 / v_rcp_f32 / v_rsq_f32).
__device__ __forceinline__ float sigmoidf_(float v) {
    return __builtin_amdgcn_rcpf(1.0f + __expf(-v));
}
__device__ __forceinline__ float geluf_(float v) {
    return 0.5f * v * (1.0f + erff(v * 0.70710678118654752f));
}

// ---------------------------------------------------------------------------
// WMMA fragment layouts (CDNA5 ISA §7.12.2, wave32):
//   A 16x32 f16 : lane l<16 row M=l,  VGPR j -> K = (j<4?0:16)+hi*8+2*(j&3)
//   B 32x16 f16 : lane n=l&15 col,    VGPR j -> K = hi*16 + 2j in this code's packing (A packed the same)
//   C/D 16x16 f32: VGPR r -> M = r + 8*hi, N = l&15
// Packed operand buffers hold, per (tile, kk-step, lane), the 16 f16 the lane
// needs, contiguously -> fragment fetch = one aligned 32B load (2x b128).
// ---------------------------------------------------------------------------

// B-fragment scatter offset for element (channel c, in-batch position p) of
// batch b, in a [globalTile][kk][lane][16] f16 buffer.
__device__ __forceinline__ size_t bswz_off(int b, int c, int p)
{
    const int pt = b * kPT + (p >> 4);  // global position tile
    const int n  = p & 15;
    const int kk = c >> 5;              // K-step (0/1)
    const int k0 = c & 31;
    const int hi = k0 >> 4;
    const int j  = (k0 & 15) >> 1;
    const int e  = k0 & 1;
    return ((size_t)(pt * 2 + kk) * 32 + (hi * 16 + n)) * 16 + 2 * j + e;
}

// Pack one 64x64 f32 weight matrix (row-major [o][c]) into A-fragment layout:
// [ot][kk][lane][16] f16.  256 threads, one (ot,kk,lane) triple per thread.
__global__ void wprep_k(const float* __restrict__ W, f16* __restrict__ Wp)
{
    const int tid  = threadIdx.x;        // 0..255
    const int ot   = tid >> 6;
    const int kkI  = (tid >> 5) & 1;
    const int lane = tid & 31;
    const int l    = lane & 15;
    const int hi   = lane >> 4;
    const int m    = ot * 16 + l;
    f16* dst = Wp + ((size_t)(ot * 2 + kkI) * 32 + lane) * 16;
#pragma unroll
    for (int j = 0; j < 8; ++j) {
        const int kb = hi * 16 + 2 * j;
        dst[2 * j]     = (f16)W[m * kC + kkI * 32 + kb];
        dst[2 * j + 1] = (f16)W[m * kC + kkI * 32 + kb + 1];
    }
}

// ---------------------------------------------------------------------------
// Pointwise (1x1) conv as WMMA GEMM:  Y[b,o,p] = act(bn(sum_c X[b,c,p]*W[o,c]))
// One wave per 16(out)x16(pos) tile, K-loop over C=64 in 2 steps of 32.
// A always from packed f16 weights (2x b128 per step).  B either from the
// packed/swizzled T0 buffer (BSWZ=true, 2x b128) or gathered from a natural
// [b][c][p] tensor (BSWZ=false; used for the harness f32 input and BUF0).
// Grid is exact => EXEC all-1s at every WMMA (ISA §7.12 requirement).
// ---------------------------------------------------------------------------
template <typename TIN, typename TOUT, bool BSWZ>
__global__ __launch_bounds__(256)
void pw_wmma_k(const TIN* __restrict__ X, const f16* __restrict__ Wp,
               const float* __restrict__ gamma, const float* __restrict__ beta,
               TOUT* __restrict__ Y, int applyBn, int act)
{
    const int wid  = blockIdx.x * 8 + (threadIdx.x >> 5);   // global wave id
    const int lane = threadIdx.x & 31;
    const int ot   = wid & 3;          // out-channel tile (4 x 16 = 64)
    const int pt   = wid >> 2;         // global position tile
    const int b    = pt / kPT;
    const int ps   = (pt % kPT) * 16;
    const int l    = lane & 15;
    const int hi   = lane >> 4;

    v8f acc = {};

#pragma unroll
    for (int kkI = 0; kkI < 2; ++kkI) {
        // A fragment: one aligned 32B load from packed weights.
        const v16h a = *(const v16h*)(Wp + ((size_t)(ot * 2 + kkI) * 32 + lane) * 16);

        v16h bb;
        if constexpr (BSWZ) {
            // B fragment: one aligned 32B load from the swizzled activation buf.
            bb = *(const v16h*)((const f16*)X +
                                ((size_t)(pt * 2 + kkI) * 32 + lane) * 16);
        } else {
            const TIN* xb = X + (size_t)b * kC * kHT;
#pragma unroll
            for (int j = 0; j < 8; ++j) {
                const int k0 = kkI * 32 + hi * 16 + 2 * j;
                bb[2 * j]     = (f16)xb[(size_t)k0       * kHT + ps + l];
                bb[2 * j + 1] = (f16)xb[(size_t)(k0 + 1) * kHT + ps + l];
            }
        }
        acc = __builtin_amdgcn_wmma_f32_16x16x32_f16(
            /*neg_a=*/false, a, /*neg_b=*/false, bb,
            /*c_mod=*/(short)0, acc, /*reuse_a=*/false, /*reuse_b=*/false);
    }

    TOUT* yb = Y + (size_t)b * kC * kHT;
#pragma unroll
    for (int r = 0; r < 8; ++r) {
        const int m = ot * 16 + r + 8 * hi;
        float v = acc[r];
        if (applyBn) v = v * gamma[m] + beta[m];
        if (act == 1)      v = fmaxf(v, 0.0f);
        else if (act == 2) v = sigmoidf_(v);
        yb[(size_t)m * kHT + ps + l] = (TOUT)v;
    }
}

// ---------------------------------------------------------------------------
// Depthwise 5x5, pad (2,2),(2,2).  f16 in, f32 accumulate, store swizzled
// into the B-fragment layout (sole consumer is pw_wmma_k<...,true>).
// ---------------------------------------------------------------------------
__global__ __launch_bounds__(256)
void dw5x5_k(const f16* __restrict__ X, const float* __restrict__ Wd,
             f16* __restrict__ Y)
{
    const long long idx = (long long)blockIdx.x * 256 + threadIdx.x;
    if (idx >= kNELEM) return;
    const int t = (int)(idx % kT);
    const int h = (int)((idx / kT) % kH);
    const int c = (int)((idx / kHT) % kC);
    const int b = (int)(idx / ((long long)kC * kHT));
    const float* w  = Wd + c * 25;
    const f16* xc = X + ((size_t)b * kC + c) * kHT;
    float s = 0.0f;
#pragma unroll
    for (int i = 0; i < 5; ++i) {
        const int hh = h + i - 2;
        if (hh < 0 || hh >= kH) continue;
#pragma unroll
        for (int j = 0; j < 5; ++j) {
            const int tt = t + j - 2;
            if (tt < 0 || tt >= kT) continue;
            s += (float)xc[hh * kT + tt] * w[i * 5 + j];
        }
    }
    Y[bswz_off(b, c, h * kT + t)] = (f16)s;
}

// ---------------------------------------------------------------------------
// Depthwise 1x3 along T, pad (0,0),(1,1).  f16 in, swizzled f16 out.
// ---------------------------------------------------------------------------
__global__ __launch_bounds__(256)
void dw1x3_k(const f16* __restrict__ X, const float* __restrict__ Wd,
             f16* __restrict__ Y)
{
    const long long idx = (long long)blockIdx.x * 256 + threadIdx.x;
    if (idx >= kNELEM) return;
    const int t = (int)(idx % kT);
    const int p = (int)(idx % kHT);
    const int c = (int)((idx / kHT) % kC);
    const int b = (int)(idx / ((long long)kC * kHT));
    const float* w = Wd + c * 3;
    float s = 0.0f;
#pragma unroll
    for (int j = 0; j < 3; ++j) {
        const int tt = t + j - 1;
        if (tt < 0 || tt >= kT) continue;
        s += (float)X[idx + (tt - t)] * w[j];
    }
    Y[bswz_off(b, c, p)] = (f16)s;
}

// ---------------------------------------------------------------------------
// Channel attention: per-(b,c) avg & max over HxT  (f16 in, f32 stats out)
// ---------------------------------------------------------------------------
__global__ __launch_bounds__(256)
void ca_reduce_k(const f16* __restrict__ X, float* __restrict__ stats)
{
    __shared__ float ssum[256];
    __shared__ float smax[256];
    const int b = blockIdx.x / kC;
    const int c = blockIdx.x % kC;
    const f16* xc = X + ((size_t)b * kC + c) * kHT;
    float s = 0.0f, m = -3.4e38f;
    for (int i = threadIdx.x; i < kHT; i += 256) {
        const float v = (float)xc[i];
        s += v;
        m = fmaxf(m, v);
    }
    ssum[threadIdx.x] = s; smax[threadIdx.x] = m;
    __syncthreads();
    for (int off = 128; off > 0; off >>= 1) {
        if (threadIdx.x < off) {
            ssum[threadIdx.x] += ssum[threadIdx.x + off];
            smax[threadIdx.x] = fmaxf(smax[threadIdx.x], smax[threadIdx.x + off]);
        }
        __syncthreads();
    }
    if (threadIdx.x == 0) {
        stats[(size_t)b * 2 * kC + c]      = ssum[0] / (float)kHT;
        stats[(size_t)b * 2 * kC + kC + c] = smax[0];
    }
}

// Tiny SE-style FC: a[b,c] = sigmoid(fc(avg) + fc(mx)), fc = relu(v@w1.T)@w2.T
__global__ void ca_fc_k(const float* __restrict__ stats,
                        const float* __restrict__ w1, const float* __restrict__ w2,
                        float* __restrict__ a)
{
    const int b = blockIdx.x;
    const int c = threadIdx.x;            // 64 threads
    const float* avg = stats + (size_t)b * 2 * kC;
    float acc = 0.0f;
#pragma unroll
    for (int src = 0; src < 2; ++src) {
        const float* v = avg + src * kC;
#pragma unroll
        for (int mid = 0; mid < kMID; ++mid) {
            float h = 0.0f;
            for (int k = 0; k < kC; ++k) h += v[k] * w1[mid * kC + k];
            h = fmaxf(h, 0.0f);
            acc += h * w2[c * kMID + mid];
        }
    }
    a[(size_t)b * kC + c] = sigmoidf_(acc);
}

// cb_pre = b1 * (b1 * a)   (in place, f16)
__global__ __launch_bounds__(256)
void ca_apply_k(f16* __restrict__ X, const float* __restrict__ a)
{
    const long long idx = (long long)blockIdx.x * 256 + threadIdx.x;
    if (idx >= kNELEM) return;
    const int c = (int)((idx / kHT) % kC);
    const int b = (int)(idx / ((long long)kC * kHT));
    const float v = (float)X[idx];
    X[idx] = (f16)(v * v * a[(size_t)b * kC + c]);
}

// ---------------------------------------------------------------------------
// Spatial attention: per-position channel mean & max -> comb[b,2,H,T] (f32)
// ---------------------------------------------------------------------------
__global__ __launch_bounds__(256)
void sa_reduce_k(const f16* __restrict__ X, float* __restrict__ comb)
{
    const int idx = blockIdx.x * 256 + threadIdx.x;
    if (idx >= kNPOS) return;
    const int b = idx / kHT;
    const int p = idx % kHT;
    const f16* xb = X + (size_t)b * kC * kHT + p;
    float s = 0.0f, m = -3.4e38f;
#pragma unroll 4
    for (int c = 0; c < kC; ++c) {
        const float v = (float)xb[(size_t)c * kHT];
        s += v;
        m = fmaxf(m, v);
    }
    comb[(size_t)b * 2 * kHT + p]       = s / (float)kC;
    comb[(size_t)b * 2 * kHT + kHT + p] = m;
}

// 7x7 conv over the 2 pooled channels, pad 3, sigmoid -> amap[b,H,T] (f32)
__global__ __launch_bounds__(256)
void sa_conv7_k(const float* __restrict__ comb, const float* __restrict__ w,
                float* __restrict__ amap)
{
    const int idx = blockIdx.x * 256 + threadIdx.x;
    if (idx >= kNPOS) return;
    const int b = idx / kHT;
    const int p = idx % kHT;
    const int h = p / kT, t = p % kT;
    float s = 0.0f;
#pragma unroll
    for (int ic = 0; ic < 2; ++ic) {
        const float* xc = comb + (size_t)b * 2 * kHT + (size_t)ic * kHT;
        for (int i = 0; i < 7; ++i) {
            const int hh = h + i - 3;
            if (hh < 0 || hh >= kH) continue;
            for (int j = 0; j < 7; ++j) {
                const int tt = t + j - 3;
                if (tt < 0 || tt >= kT) continue;
                s += xc[hh * kT + tt] * w[ic * 49 + i * 7 + j];
            }
        }
    }
    amap[(size_t)b * kHT + p] = sigmoidf_(s);
}

// sb = sigmoid(b2 * (b2 * amap))   (in place, f16)
__global__ __launch_bounds__(256)
void sa_apply_k(f16* __restrict__ X, const float* __restrict__ amap)
{
    const long long idx = (long long)blockIdx.x * 256 + threadIdx.x;
    if (idx >= kNELEM) return;
    const int p = (int)(idx % kHT);
    const int b = (int)(idx / ((long long)kC * kHT));
    const float v = (float)X[idx];
    X[idx] = (f16)sigmoidf_(v * v * amap[(size_t)b * kHT + p]);
}

// ---------------------------------------------------------------------------
// SSM: state update never reads the input => state_t is identical for every
// (b,h) row. Precompute y[t][c] = s_t @ Cm once (1 block, 400 serial steps).
// ---------------------------------------------------------------------------
__global__ void ssm_pre_k(const float* __restrict__ A, const float* __restrict__ Bv,
                          const float* __restrict__ Cm, float* __restrict__ Yt)
{
    __shared__ float s[kS];
    __shared__ float sn[kS];
    const int tid = threadIdx.x;   // 64 threads
    if (tid < kS) s[tid] = 0.0f;
    __syncthreads();
    for (int t = 0; t < kT; ++t) {
        if (tid < kS) {
            float acc = Bv[tid];
#pragma unroll
            for (int j = 0; j < kS; ++j) acc += s[j] * A[tid * kS + j]; // (s @ A.T)[i]
            sn[tid] = acc;
        }
        __syncthreads();
        if (tid < kS) s[tid] = sn[tid];
        __syncthreads();
        float y = 0.0f;
#pragma unroll
        for (int j = 0; j < kS; ++j) y += s[j] * Cm[j * kC + tid];
        Yt[t * kC + tid] = y;
        __syncthreads();
    }
}

// Fused: u = x + gelu(y[t] + x*D);  LayerNorm over channels;  * sb;
// store with T-flip (o[..., ::-1]) folded into the write.  f16 in/out.
__global__ __launch_bounds__(256)
void ssm_ln_fused_k(const f16* __restrict__ CB, const f16* __restrict__ SB,
                    const float* __restrict__ Yt, const float* __restrict__ D,
                    const float* __restrict__ lng, const float* __restrict__ lnb,
                    f16* __restrict__ O)
{
    const int idx = blockIdx.x * 256 + threadIdx.x;
    if (idx >= kNPOS) return;
    const int b = idx / kHT;
    const int p = idx % kHT;
    const int t = p % kT;
    const int h = p / kT;
    const f16* cbb = CB + (size_t)b * kC * kHT + p;
    const float* yt = Yt + t * kC;

    float mean = 0.0f, m2 = 0.0f;
#pragma unroll 4
    for (int c = 0; c < kC; ++c) {
        const float x = (float)cbb[(size_t)c * kHT];
        const float u = x + geluf_(yt[c] + x * D[c]);
        mean += u; m2 += u * u;
    }
    mean *= (1.0f / kC);
    const float var = m2 * (1.0f / kC) - mean * mean;
    const float inv = __builtin_amdgcn_rsqf(var + 1e-5f);   // native v_rsq_f32

    const f16* sbb = SB + (size_t)b * kC * kHT + p;
    f16* ob = O + (size_t)b * kC * kHT + (size_t)h * kT + (kT - 1 - t);
#pragma unroll 4
    for (int c = 0; c < kC; ++c) {
        const float x = (float)cbb[(size_t)c * kHT];
        const float u = x + geluf_(yt[c] + x * D[c]);
        const float n = (u - mean) * inv * lng[c] + lnb[c];
        ob[(size_t)c * kHT] = (f16)(n * (float)sbb[(size_t)c * kHT]);
    }
}

// ---------------------------------------------------------------------------
// Host-side orchestration
// ---------------------------------------------------------------------------
extern "C" void kernel_launch(void* const* d_in, const int* in_sizes, int n_in,
                              void* d_out, int out_size, void* d_ws, size_t ws_size,
                              hipStream_t stream)
{
    const float* x       = (const float*)d_in[0];
    const float* in_w    = (const float*)d_in[1];
    const float* dw2d_dw = (const float*)d_in[2];
    const float* dw2d_pw = (const float*)d_in[3];
    const float* dw2d_g  = (const float*)d_in[4];
    const float* dw2d_b  = (const float*)d_in[5];
    const float* ca_w1   = (const float*)d_in[6];
    const float* ca_w2   = (const float*)d_in[7];
    const float* sa_w    = (const float*)d_in[8];
    const float* dw1d_dw = (const float*)d_in[9];
    const float* dw1d_pw = (const float*)d_in[10];
    const float* dw1d_g  = (const float*)d_in[11];
    const float* dw1d_b  = (const float*)d_in[12];
    const float* ssm_A   = (const float*)d_in[13];
    const float* ssm_B   = (const float*)d_in[14];
    const float* ssm_C   = (const float*)d_in[15];
    const float* ssm_D   = (const float*)d_in[16];
    const float* ln_g    = (const float*)d_in[17];
    const float* ln_b    = (const float*)d_in[18];
    const float* out_w   = (const float*)d_in[19];
    const float* out_g   = (const float*)d_in[20];
    const float* out_b   = (const float*)d_in[21];

    // Workspace: 4 f16 ping-pong activation buffers (~102 MB total -> resident
    // in the 192 MB L2) + small f32 tensors + packed f16 weights.
    f16* wsh = (f16*)d_ws;
    f16* BUF0 = wsh;                        // main activation / half output
    f16* T0   = BUF0 + kNELEM;              // GEMM input scratch (B-swizzled)
    f16* B1   = T0   + kNELEM;              // branch 1 -> cb
    f16* B2   = B1   + kNELEM;              // branch 2 -> sb
    float* smallf = (float*)(B2 + kNELEM);
    float* stats = smallf;                        // B*2*C
    float* caa   = stats + (size_t)kB * 2 * kC;   // B*C
    float* yt    = caa   + (size_t)kB * kC;       // T*C
    float* comb  = yt    + (size_t)kT * kC;       // B*2*H*T
    float* amap  = comb  + (size_t)kB * 2 * kHT;  // B*H*T
    f16*  Wp     = (f16*)(amap + (size_t)kB * kHT); // 8 x 4096 f16 (64 KB)

    const dim3 blk(256);
    const int gElem = (int)(kNELEM / 256);        // 49600
    const int gPos  = kNPOS / 256;                // 775
    const int gGemm = (4 * (kNPOS / 16)) / 8;     // 6200 blocks x 8 waves

    // Pre-pack all 8 weight matrices into A-fragment layout (once per call).
    wprep_k<<<1, 256, 0, stream>>>(in_w,                          Wp + 0 * kWSLOT);
    wprep_k<<<1, 256, 0, stream>>>(dw2d_pw + 0 * kC * kC,         Wp + 1 * kWSLOT);
    wprep_k<<<1, 256, 0, stream>>>(dw2d_pw + 1 * kC * kC,         Wp + 2 * kWSLOT);
    wprep_k<<<1, 256, 0, stream>>>(dw1d_pw + 0 * kC * kC,         Wp + 3 * kWSLOT);
    wprep_k<<<1, 256, 0, stream>>>(dw2d_pw + 2 * kC * kC,         Wp + 4 * kWSLOT);
    wprep_k<<<1, 256, 0, stream>>>(dw2d_pw + 3 * kC * kC,         Wp + 5 * kWSLOT);
    wprep_k<<<1, 256, 0, stream>>>(dw1d_pw + 1 * kC * kC,         Wp + 6 * kWSLOT);
    wprep_k<<<1, 256, 0, stream>>>(out_w,                         Wp + 7 * kWSLOT);

    // h = pw(x, in_w)                                   x(f32) -> BUF0(f16)
    pw_wmma_k<float, f16, false><<<gGemm, blk, 0, stream>>>(
        x, Wp + 0 * kWSLOT, nullptr, nullptr, BUF0, 0, 0);

    for (int i = 0; i < 2; ++i) {
        const float* dwA = dw2d_dw + (size_t)(i * 2 + 0) * kC * 25;
        const float* dwB = dw2d_dw + (size_t)(i * 2 + 1) * kC * 25;
        const f16* pwA = Wp + (size_t)(i * 3 + 1) * kWSLOT;
        const f16* pwB = Wp + (size_t)(i * 3 + 2) * kWSLOT;
        const f16* pwD = Wp + (size_t)(i * 3 + 3) * kWSLOT;
        const float* gA  = dw2d_g + (size_t)(i * 2 + 0) * kC;
        const float* bA  = dw2d_b + (size_t)(i * 2 + 0) * kC;
        const float* gB  = dw2d_g + (size_t)(i * 2 + 1) * kC;
        const float* bB  = dw2d_b + (size_t)(i * 2 + 1) * kC;

        // b1 = relu(bn(pw(dw5x5(x))))                  BUF0 -> T0(swz) -> B1
        dw5x5_k<<<gElem, blk, 0, stream>>>(BUF0, dwA, T0);
        pw_wmma_k<f16, f16, true><<<gGemm, blk, 0, stream>>>(T0, pwA, gA, bA, B1, 1, 1);
        // b2 = relu(bn(pw(dw5x5(x))))                  BUF0 -> T0(swz) -> B2
        dw5x5_k<<<gElem, blk, 0, stream>>>(BUF0, dwB, T0);
        pw_wmma_k<f16, f16, true><<<gGemm, blk, 0, stream>>>(T0, pwB, gB, bB, B2, 1, 1);

        // channel attention: cb_pre = b1*b1*a          (B1 in place)
        ca_reduce_k<<<kB * kC, blk, 0, stream>>>(B1, stats);
        ca_fc_k   <<<kB, 64, 0, stream>>>(stats, ca_w1 + (size_t)i * kMID * kC,
                                          ca_w2 + (size_t)i * kC * kMID, caa);
        ca_apply_k<<<gElem, blk, 0, stream>>>(B1, caa);

        // cb = sigmoid(bn(pw(dw1x3(cb_pre))))          B1 -> T0(swz) -> B1
        dw1x3_k<<<gElem, blk, 0, stream>>>(B1, dw1d_dw + (size_t)i * kC * 3, T0);
        pw_wmma_k<f16, f16, true><<<gGemm, blk, 0, stream>>>(
            T0, pwD, dw1d_g + (size_t)i * kC, dw1d_b + (size_t)i * kC, B1, 1, 2);

        // sb = sigmoid(b2*b2*sigmoid(conv7(pool(b2)))) (B2 in place)
        sa_reduce_k<<<gPos, blk, 0, stream>>>(B2, comb);
        sa_conv7_k <<<gPos, blk, 0, stream>>>(comb, sa_w + (size_t)i * 2 * 49, amap);
        sa_apply_k <<<gElem, blk, 0, stream>>>(B2, amap);

        // SSM (input-independent state) + GELU + residual + LN + *sb, T-flip
        ssm_pre_k     <<<1, 64, 0, stream>>>(ssm_A + (size_t)i * kS * kS,
                                             ssm_B + (size_t)i * kS,
                                             ssm_C + (size_t)i * kS * kC, yt);
        ssm_ln_fused_k<<<gPos, blk, 0, stream>>>(B1, B2, yt,
                                                 ssm_D + (size_t)i * kC,
                                                 ln_g + (size_t)i * kC,
                                                 ln_b + (size_t)i * kC, BUF0);
    }

    // out = relu(bn(pw(o, out_w)))                     BUF0(f16) -> d_out(f32)
    pw_wmma_k<f16, float, false><<<gGemm, blk, 0, stream>>>(
        BUF0, Wp + 7 * kWSLOT, out_g, out_b, (float*)d_out, 1, 1);
}
